// ManyToManyLSTMM_1322849927225
// MI455X (gfx1250) — matrix operand order repaired and checked by reading, not compiled, not run on hardware
//
#include <hip/hip_runtime.h>
#include <stdint.h>

#define H      1024
#define FOURH  4096
#define BATCH  64
#define SEQ    512
#define RBLOCKS 128   // persistent recurrent blocks; 128 blocks * 32 cols = 4096

typedef __attribute__((ext_vector_type(16))) __bf16 bf16x16;
typedef __attribute__((ext_vector_type(8)))  float  f32x8;
typedef unsigned int u32x4 __attribute__((ext_vector_type(4)));
typedef int          i32x8 __attribute__((ext_vector_type(8)));
typedef int          i32x4 __attribute__((ext_vector_type(4)));

union Frag { bf16x16 v; uint4 q[2]; };

__device__ __forceinline__ unsigned short f32_to_bf16(float f) {
    union { float f; unsigned u; } x; x.f = f;
    unsigned u = x.u;
    unsigned r = u + 0x7FFFu + ((u >> 16) & 1u);   // round-to-nearest-even
    if ((u & 0x7F800000u) == 0x7F800000u) r = u;   // inf/nan passthrough
    return (unsigned short)(r >> 16);
}
__device__ __forceinline__ float bf16_to_f32(unsigned short h) {
    union { unsigned u; float f; } x; x.u = ((unsigned)h) << 16;
    return x.f;
}

// Load A (16x32 of h, row-major) and B (32x16 = 16 rows of W, row-major)
// fragments per ISA 7.12.2 layouts and issue one WMMA into *acc.
__device__ __forceinline__ void wmma_step(const unsigned short* arow,
                                          const unsigned short* brow,
                                          int kt, int hlf, f32x8* acc) {
    Frag a, b;
    a.q[0] = *(const uint4*)(arow + kt + hlf * 8);
    a.q[1] = *(const uint4*)(arow + kt + 16 + hlf * 8);
    b.q[0] = *(const uint4*)(brow + kt + hlf * 8);
    b.q[1] = *(const uint4*)(brow + kt + 16 + hlf * 8);
    *acc = __builtin_amdgcn_wmma_f32_16x16x32_bf16(false, a.v, false, b.v,
                                                   (short)0, *acc, false, false);
}

// ---------------------------------------------------------------------------
// Kernel 1: build GATE-INTERLEAVED bf16 weights (row n' = hidx*4 + gate maps
// to original row gate*H + hidx), combined interleaved bias, bf16 x, h_bf16.
// ---------------------------------------------------------------------------
__global__ void prep_kernel(const float* __restrict__ x,
                            const float* __restrict__ hx0,
                            const float* __restrict__ W_ih,
                            const float* __restrict__ W_hh,
                            const float* __restrict__ b_ih,
                            const float* __restrict__ b_hh,
                            unsigned short* __restrict__ wih_p,
                            unsigned short* __restrict__ whh_p,
                            unsigned short* __restrict__ x_bf,
                            float* __restrict__ bias_p,
                            unsigned short* __restrict__ h_bf,
                            unsigned* __restrict__ bar)
{
    const size_t NW  = (size_t)FOURH * H;
    const size_t NX  = (size_t)SEQ * BATCH * H;
    const size_t NBH = (size_t)BATCH * H;
    const size_t total = NW * 2 + NX + FOURH + NBH;
    for (size_t i = (size_t)blockIdx.x * blockDim.x + threadIdx.x; i < total;
         i += (size_t)gridDim.x * blockDim.x) {
        size_t j = i;
        if (j < NW) {  // permuted W_ih
            int np = (int)(j >> 10), k = (int)(j & (H - 1));
            int hidx = np >> 2, g = np & 3;
            wih_p[j] = f32_to_bf16(W_ih[(size_t)(g * H + hidx) * H + k]);
            continue;
        }
        j -= NW;
        if (j < NW) {  // permuted W_hh
            int np = (int)(j >> 10), k = (int)(j & (H - 1));
            int hidx = np >> 2, g = np & 3;
            whh_p[j] = f32_to_bf16(W_hh[(size_t)(g * H + hidx) * H + k]);
            continue;
        }
        j -= NW;
        if (j < NX) { x_bf[j] = f32_to_bf16(x[j]); continue; }
        j -= NX;
        if (j < FOURH) {  // permuted bias
            int np = (int)j, hidx = np >> 2, g = np & 3;
            bias_p[np] = b_ih[g * H + hidx] + b_hh[g * H + hidx];
            continue;
        }
        j -= FOURH;
        h_bf[j] = f32_to_bf16(hx0[j]);
    }
    if (blockIdx.x == 0 && threadIdx.x < 2) bar[threadIdx.x] = 0;
}

// ---------------------------------------------------------------------------
// Kernel 2: gxp[s,b,n'] = bf16( bias_p[n'] + sum_k x[s,b,k] * Wih_p[n',k] )
// One 16x16 tile per wave; 4 independent accumulator chains over K=1024.
// ---------------------------------------------------------------------------
__global__ __launch_bounds__(256) void gemm_x_kernel(
    const unsigned short* __restrict__ x_bf,
    const unsigned short* __restrict__ wih_p,
    const float* __restrict__ bias_p,
    unsigned short* __restrict__ gxp)
{
    const int lane  = threadIdx.x & 31;
    const int wave  = threadIdx.x >> 5;
    const int tile  = blockIdx.x * 8 + wave;       // 2048 * 256 tiles
    const int mtile = tile >> 8;
    const int ntile = tile & 255;
    const int hlf   = lane >> 4;
    const int mrow  = mtile * 16 + (lane & 15);
    const int ncol  = ntile * 16 + (lane & 15);

    const unsigned short* arow = x_bf  + (size_t)mrow * H;
    const unsigned short* brow = wih_p + (size_t)ncol * H;

    f32x8 a0 = {}, a1 = {}, a2 = {}, a3 = {};
    #pragma unroll 2
    for (int kt = 0; kt < H; kt += 128) {
        wmma_step(arow, brow, kt,      hlf, &a0);
        wmma_step(arow, brow, kt + 32, hlf, &a1);
        wmma_step(arow, brow, kt + 64, hlf, &a2);
        wmma_step(arow, brow, kt + 96, hlf, &a3);
    }
    f32x8 acc = (a0 + a1) + (a2 + a3);

    unsigned short* out = gxp + (size_t)(mtile * 16) * FOURH + ncol;
    const float bb = bias_p[ncol];
    #pragma unroll
    for (int v = 0; v < 8; ++v) {
        int m = v + hlf * 8;
        out[(size_t)m * FOURH] = f32_to_bf16(acc[v] + bb);
    }
}

// ---------------------------------------------------------------------------
// Device-wide sense-reversing barrier
// ---------------------------------------------------------------------------
__device__ __forceinline__ void grid_sync(unsigned* cnt, unsigned* gen) {
    __threadfence();
    __syncthreads();
    if (threadIdx.x == 0) {
        unsigned g0 = __hip_atomic_load(gen, __ATOMIC_RELAXED, __HIP_MEMORY_SCOPE_AGENT);
        unsigned prev = __hip_atomic_fetch_add(cnt, 1u, __ATOMIC_ACQ_REL, __HIP_MEMORY_SCOPE_AGENT);
        if (prev == RBLOCKS - 1) {
            __hip_atomic_store(cnt, 0u, __ATOMIC_RELAXED, __HIP_MEMORY_SCOPE_AGENT);
            __hip_atomic_fetch_add(gen, 1u, __ATOMIC_RELEASE, __HIP_MEMORY_SCOPE_AGENT);
        } else {
            while (__hip_atomic_load(gen, __ATOMIC_ACQUIRE, __HIP_MEMORY_SCOPE_AGENT) == g0)
                __builtin_amdgcn_s_sleep(1);
        }
    }
    __syncthreads();
    __threadfence();
}

// ---------------------------------------------------------------------------
// Dynamic LDS layout for the recurrent kernel (~84 KB / block)
// ---------------------------------------------------------------------------
struct SMem {
    unsigned short w[32 * H];          // 64 KB  W_hh_p rows [32*blk, +32)
    float          gbuf[64 * 32];      //  8 KB  recurrent gate pre-activations
    unsigned short gxbuf[2][64 * 32];  //  8 KB  double-buffered gatesX tile (TDM dest)
    float          cbuf[64 * 8];       //  2 KB  cell state slice (block-owned)
    float          biasb[32];          // 128 B  bias slice
};

// ---------------------------------------------------------------------------
// Kernel 3: persistent recurrent loop; ONE grid sync per step.
// Block b owns output columns n' in [32b, 32b+32) == hidden [8b, 8b+8) x 4 gates.
// TDM prefetches next step's gatesX tile into LDS (TENSORcnt-tracked).
// ---------------------------------------------------------------------------
__global__ __launch_bounds__(256) void lstm_recurrent_kernel(
    const unsigned short* __restrict__ whh_p,
    const unsigned short* __restrict__ wih_p,
    const unsigned short* __restrict__ gxp,
    const float* __restrict__ bias_p,
    const float* __restrict__ cx0,
    unsigned short* __restrict__ h_bf,
    float* __restrict__ out,
    unsigned* __restrict__ bar)
{
    extern __shared__ char smem_raw[];
    SMem* sm = (SMem*)smem_raw;

    const int lane  = threadIdx.x & 31;
    const int wave  = threadIdx.x >> 5;
    const int hlf   = lane >> 4;
    const int mtile = wave & 3;              // 4 M-tiles cover B=64
    const int ntl   = wave >> 2;             // local N-tile 0/1
    const int mrow  = mtile * 16 + (lane & 15);
    const int ncol  = blockIdx.x * 32 + ntl * 16 + (lane & 15);
    const int hbase = blockIdx.x * 8;        // block-owned hidden range

    // ---- init: W_hh_p slice -> LDS; c slice -> LDS; bias slice -> LDS ----
    {
        const uint4* src = (const uint4*)(whh_p + (size_t)blockIdx.x * 32 * H);
        uint4* dst = (uint4*)sm->w;
        for (int i = threadIdx.x; i < (32 * H * 2) / 16; i += 256) dst[i] = src[i];
        for (int e = threadIdx.x; e < BATCH * 8; e += 256) {
            int bb = e >> 3, hl = e & 7;
            sm->cbuf[e] = cx0[(size_t)bb * H + hbase + hl];
        }
        if (threadIdx.x < 32) sm->biasb[threadIdx.x] = bias_p[blockIdx.x * 32 + threadIdx.x];
    }
    __syncthreads();

    // ---- TDM descriptor (constant part) for a 64x32-elem bf16 tile,
    //      row stride 4096 elements, out of gxp ----
    i32x8 tg1;
    tg1[0] = (1 << 16);                         // data_size = 1 (2 bytes/elem)
    tg1[1] = (int)((FOURH & 0xFFFF) << 16);     // tensor_dim0 = 4096 (lo16)
    tg1[2] = (int)((FOURH >> 16) | ((0x100000 & 0xFFFF) << 16)); // dim0 hi / dim1 lo
    tg1[3] = (int)((0x100000 >> 16) | (32 << 16));   // dim1 hi | tile_dim0 = 32
    tg1[4] = 64;                                // tile_dim1 = 64 (tile_dim2 = 0)
    tg1[5] = FOURH;                             // tensor_dim0_stride = 4096
    tg1[6] = 0;
    tg1[7] = 0;
    const i32x4 zero4 = {};
    const i32x8 zero8 = {};
    const unsigned lds_gx0 = (unsigned)((char*)&sm->gxbuf[0][0] - smem_raw);
    const unsigned lds_gx1 = (unsigned)((char*)&sm->gxbuf[1][0] - smem_raw);

    // prologue: issue TDM load of step 0's tile into gxbuf[0]
    if (wave == 0) {
        uint64_t ga = (uint64_t)(uintptr_t)(gxp + (size_t)32 * blockIdx.x);
        u32x4 g0;
        g0[0] = 1u;                                  // count = 1
        g0[1] = lds_gx0;                             // lds_addr
        g0[2] = (unsigned)(ga & 0xFFFFFFFFu);        // global_addr lo
        g0[3] = (unsigned)((ga >> 32) & 0x01FFFFFFu) | 0x80000000u; // hi | type=2
        __builtin_amdgcn_tensor_load_to_lds(g0, tg1, zero4, zero4, zero8, 0);
    }

    unsigned* cnt = bar;
    unsigned* gen = bar + 1;
    const unsigned short* bldss = sm->w + (size_t)(ntl * 16 + (lane & 15)) * H;

    for (int step = 0; step < SEQ + 2; ++step) {
        // ---- recurrent GEMM tile: acc = h @ W_hh_p^T  (4 independent chains) ----
        const unsigned short* arow = h_bf + (size_t)mrow * H;
        f32x8 a0 = {}, a1 = {}, a2 = {}, a3 = {};
        #pragma unroll 2
        for (int kt = 0; kt < H; kt += 128) {
            wmma_step(arow, bldss, kt,      hlf, &a0);
            wmma_step(arow, bldss, kt + 32, hlf, &a1);
            wmma_step(arow, bldss, kt + 64, hlf, &a2);
            wmma_step(arow, bldss, kt + 96, hlf, &a3);
        }
        if (step >= SEQ) {   // autoregressive: input is h -> also h @ W_ih_p^T
            const unsigned short* brow = wih_p + (size_t)ncol * H;
            #pragma unroll 2
            for (int kt = 0; kt < H; kt += 128) {
                wmma_step(arow, brow, kt,      hlf, &a0);
                wmma_step(arow, brow, kt + 32, hlf, &a1);
                wmma_step(arow, brow, kt + 64, hlf, &a2);
                wmma_step(arow, brow, kt + 96, hlf, &a3);
            }
        }
        f32x8 acc = (a0 + a1) + (a2 + a3);

        // issue TDM prefetch for step+1 while results drain to LDS
        if (wave == 0 && step + 1 < SEQ) {
            uint64_t ga = (uint64_t)(uintptr_t)
                (gxp + ((size_t)(step + 1) * BATCH) * FOURH + 32 * blockIdx.x);
            u32x4 g0;
            g0[0] = 1u;
            g0[1] = ((step + 1) & 1) ? lds_gx1 : lds_gx0;
            g0[2] = (unsigned)(ga & 0xFFFFFFFFu);
            g0[3] = (unsigned)((ga >> 32) & 0x01FFFFFFu) | 0x80000000u;
            __builtin_amdgcn_tensor_load_to_lds(g0, tg1, zero4, zero4, zero8, 0);
        }

        // write gate pre-activations to LDS (C layout: M = v + 8*hlf)
        #pragma unroll
        for (int v = 0; v < 8; ++v) {
            int m = mtile * 16 + v + hlf * 8;
            sm->gbuf[m * 32 + ntl * 16 + (lane & 15)] = acc[v];
        }

        // wait for this step's gatesX tile (loads complete in order)
        if (wave == 0 && step < SEQ) {
            if (step + 1 < SEQ) __builtin_amdgcn_s_wait_tensorcnt(1);
            else                __builtin_amdgcn_s_wait_tensorcnt(0);
        }
        __syncthreads();

        // ---- block-local elementwise: 64 batches x 8 hidden ----
        const unsigned short* gxt = sm->gxbuf[step & 1];
        for (int e = threadIdx.x; e < BATCH * 8; e += 256) {
            const int bb = e >> 3, hl = e & 7;
            const float* grow = sm->gbuf + bb * 32 + hl * 4;
            float gi, gf, gg, go;
            if (step < SEQ) {
                const unsigned short* gx = gxt + bb * 32 + hl * 4;
                gi = grow[0] + bf16_to_f32(gx[0]);
                gf = grow[1] + bf16_to_f32(gx[1]);
                gg = grow[2] + bf16_to_f32(gx[2]);
                go = grow[3] + bf16_to_f32(gx[3]);
            } else {
                gi = grow[0] + sm->biasb[hl * 4 + 0];
                gf = grow[1] + sm->biasb[hl * 4 + 1];
                gg = grow[2] + sm->biasb[hl * 4 + 2];
                go = grow[3] + sm->biasb[hl * 4 + 3];
            }
            const float si = 1.0f / (1.0f + __expf(-gi));
            const float sf = 1.0f / (1.0f + __expf(-gf));
            const float so = 1.0f / (1.0f + __expf(-go));
            const float tg = tanhf(gg);
            const float cn = sf * sm->cbuf[e] + si * tg;
            const float hn = so * tanhf(cn);
            sm->cbuf[e] = cn;
            h_bf[(size_t)bb * H + hbase + hl] = f32_to_bf16(hn);
            if (step >= SEQ)
                out[(size_t)(step - SEQ) * (BATCH * H) + (size_t)bb * H + hbase + hl] = hn;
        }
        grid_sync(cnt, gen);   // publish h before anyone starts the next GEMM
    }
}

// ---------------------------------------------------------------------------
extern "C" void kernel_launch(void* const* d_in, const int* in_sizes, int n_in,
                              void* d_out, int out_size, void* d_ws, size_t ws_size,
                              hipStream_t stream)
{
    const float* x    = (const float*)d_in[0];
    const float* hx0  = (const float*)d_in[1];
    const float* cx0  = (const float*)d_in[2];
    const float* W_ih = (const float*)d_in[3];
    const float* W_hh = (const float*)d_in[4];
    const float* b_ih = (const float*)d_in[5];
    const float* b_hh = (const float*)d_in[6];

    char* ws = (char*)d_ws;
    const size_t OFF_WIH  = 0;                                         // 8 MB
    const size_t OFF_WHH  = OFF_WIH + (size_t)FOURH * H * 2;           // 8 MB
    const size_t OFF_XBF  = OFF_WHH + (size_t)FOURH * H * 2;           // 64 MB
    const size_t OFF_GX   = OFF_XBF + (size_t)SEQ * BATCH * H * 2;     // 256 MB
    const size_t OFF_HBF  = OFF_GX  + (size_t)SEQ * BATCH * FOURH * 2; // 128 KB
    const size_t OFF_BIAS = OFF_HBF + (size_t)BATCH * H * 2;           // 16 KB
    const size_t OFF_BAR  = OFF_BIAS + (size_t)FOURH * 4;              // 8 B

    unsigned short* wih_p  = (unsigned short*)(ws + OFF_WIH);
    unsigned short* whh_p  = (unsigned short*)(ws + OFF_WHH);
    unsigned short* x_bf   = (unsigned short*)(ws + OFF_XBF);
    unsigned short* gxp    = (unsigned short*)(ws + OFF_GX);
    unsigned short* h_bf   = (unsigned short*)(ws + OFF_HBF);
    float*          bias_p = (float*)(ws + OFF_BIAS);
    unsigned*       bar    = (unsigned*)(ws + OFF_BAR);

    prep_kernel<<<2048, 256, 0, stream>>>(x, hx0, W_ih, W_hh, b_ih, b_hh,
                                          wih_p, whh_p, x_bf, bias_p, h_bf, bar);

    gemm_x_kernel<<<65536, 256, 0, stream>>>(x_bf, wih_p, bias_p, gxp);

    lstm_recurrent_kernel<<<RBLOCKS, 256, sizeof(SMem), stream>>>(
        whh_p, wih_p, gxp, bias_p, cx0, h_bf, (float*)d_out, bar);
}
